// MoDBlock_53068615909661
// MI455X (gfx1250) — matrix operand order repaired and checked by reading, not compiled, not run on hardware
//
#include <hip/hip_runtime.h>
#include <hip/hip_bf16.h>

typedef __bf16 bf16;
typedef __attribute__((ext_vector_type(16))) __bf16 v16bf;
typedef __attribute__((ext_vector_type(8)))  float  v8f;
typedef int v4i __attribute__((ext_vector_type(4)));

#define B_   4
#define S_   4096
#define D_   1024
#define F_   4096
#define KCAP 2048
#define M_   (B_ * KCAP)   // 8192 selected tokens

#define BM 128
#define BN 128
#define BK 32
#define ATS 40   // LDS row stride (elements), 80B: keeps 16B alignment, spreads banks
#define BTS 40

// ---------------- CDNA5 async global->LDS copy (ASYNCcnt) with sync fallback ----------------
#if __has_builtin(__builtin_amdgcn_global_load_async_to_lds_b128)
__device__ __forceinline__ void cp16_async(const void* g, void* l) {
    __builtin_amdgcn_global_load_async_to_lds_b128(
        (__attribute__((address_space(1))) v4i*)g,
        (__attribute__((address_space(3)))  v4i*)l, 0, 0);
}
#else
__device__ __forceinline__ void cp16_async(const void* g, void* l) {
    *(uint4*)l = *(const uint4*)g;   // sync fallback: still clean b128 copies
}
#endif

#if __has_builtin(__builtin_amdgcn_s_wait_asynccnt)
#define WAIT_ASYNC(n) __builtin_amdgcn_s_wait_asynccnt(n)
#else
#define WAIT_ASYNC(n) asm volatile("s_wait_asynccnt %0" :: "n"(n) : "memory")
#endif

// ---------------------------------------------------------------- scores
__global__ __launch_bounds__(256)
void mod_scores_kernel(const float* __restrict__ h, const float* __restrict__ wr,
                       float* __restrict__ scores) {
    int row  = blockIdx.x * (blockDim.x >> 5) + (threadIdx.x >> 5);
    int lane = threadIdx.x & 31;
    const float* hp = h + (size_t)row * D_;
    float sum = 0.f;
#pragma unroll
    for (int base = 0; base < D_; base += 128) {
        float4 v = *(const float4*)(hp + base + lane * 4);
        float4 w = *(const float4*)(wr + base + lane * 4);
        sum += v.x * w.x + v.y * w.y + v.z * w.z + v.w * w.w;
    }
#pragma unroll
    for (int off = 16; off > 0; off >>= 1) sum += __shfl_down(sum, off, 32);
    if (lane == 0) scores[row] = sum;
}

// ---------------------------------------------------------------- exact top-k (bitonic in LDS)
__global__ __launch_bounds__(1024)
void mod_topk_kernel(const float* __restrict__ scores,
                     int* __restrict__ top_idx, float* __restrict__ weights) {
    __shared__ float skey[S_];
    __shared__ int   sidx[S_];
    int b = blockIdx.x;
    const float* sc = scores + (size_t)b * S_;
    for (int i = threadIdx.x; i < S_; i += blockDim.x) { skey[i] = sc[i]; sidx[i] = i; }
    __syncthreads();
    for (int sz = 2; sz <= S_; sz <<= 1) {
        for (int stride = sz >> 1; stride > 0; stride >>= 1) {
            for (int i = threadIdx.x; i < S_; i += blockDim.x) {
                int j = i ^ stride;
                if (j > i) {
                    bool desc = ((i & sz) == 0);
                    float ki = skey[i], kj = skey[j];
                    bool sw = desc ? (ki < kj) : (ki > kj);
                    if (sw) {
                        skey[i] = kj; skey[j] = ki;
                        int t = sidx[i]; sidx[i] = sidx[j]; sidx[j] = t;
                    }
                }
            }
            __syncthreads();
        }
    }
    for (int i = threadIdx.x; i < KCAP; i += blockDim.x) {
        top_idx[b * KCAP + i] = sidx[i];
        weights[b * KCAP + i] = 1.0f / (1.0f + __expf(-skey[i]));
    }
}

// ---------------------------------------------------------------- gather selected -> bf16
__global__ __launch_bounds__(256)
void mod_gather_kernel(const float* __restrict__ h, const int* __restrict__ top_idx,
                       bf16* __restrict__ X) {
    int row = blockIdx.x;
    int b   = row >> 11;               // / KCAP
    int s   = top_idx[row];
    const float* src = h + ((size_t)b * S_ + s) * D_;
    bf16* dst = X + (size_t)row * D_;
    for (int d = threadIdx.x * 4; d < D_; d += blockDim.x * 4) {
        float4 v = *(const float4*)(src + d);
        union { bf16 o[4]; uint2 u; } t;
        t.o[0] = (bf16)v.x; t.o[1] = (bf16)v.y; t.o[2] = (bf16)v.z; t.o[3] = (bf16)v.w;
        *(uint2*)(dst + d) = t.u;
    }
}

// -------------------------------- f32 -> bf16 transpose-convert: out[n*K+k] = in[k*N+n]
__global__ __launch_bounds__(256)
void mod_cvt_t_kernel(const float* __restrict__ in, bf16* __restrict__ out, int K, int N) {
    __shared__ float tile[32][33];
    int k0 = blockIdx.y * 32, n0 = blockIdx.x * 32;
    int tx = threadIdx.x & 31, ty = threadIdx.x >> 5;
#pragma unroll
    for (int r = ty; r < 32; r += 8)
        tile[r][tx] = in[(size_t)(k0 + r) * N + n0 + tx];
    __syncthreads();
#pragma unroll
    for (int r = ty; r < 32; r += 8)
        out[(size_t)(n0 + r) * K + k0 + tx] = (bf16)tile[tx][r];
}

// ---------------------------------------------------------------- WMMA fragments
union FragU { v16bf v; uint4 q[2]; };

__device__ __forceinline__ void load_a_frag(FragU& f, const bf16* At, int row, int half) {
    const bf16* p = At + row * ATS + half * 8;
    f.q[0] = *(const uint4*)(p);
    f.q[1] = *(const uint4*)(p + 16);
}
__device__ __forceinline__ void load_b_frag(FragU& f, const bf16* Bt, int col, int half) {
    const bf16* p = Bt + col * BTS + half * 8;
    f.q[0] = *(const uint4*)(p);
    f.q[1] = *(const uint4*)(p + 16);
}

// stage one (BM x BK) A tile and (BN x BK) B tile (both K-contiguous) into LDS
__device__ __forceinline__ void stage_tiles(const bf16* __restrict__ A, int lda,
                                            const bf16* __restrict__ Bt, int ldb,
                                            bf16* Abuf, bf16* Bbuf,
                                            int m0, int n0, int k0, int t) {
#pragma unroll
    for (int i = 0; i < 2; ++i) {                 // A: 128 rows * 4 chunks = 512 chunks
        int c = t + i * 256;
        int row = c >> 2, kc = (c & 3) * 8;
        cp16_async(A + (size_t)(m0 + row) * lda + k0 + kc, Abuf + row * ATS + kc);
    }
#pragma unroll
    for (int i = 0; i < 2; ++i) {                 // B: 128 rows * 4 chunks = 512 chunks
        int c = t + i * 256;
        int row = c >> 2, kc = (c & 3) * 8;
        cp16_async(Bt + (size_t)(n0 + row) * ldb + k0 + kc, Bbuf + row * BTS + kc);
    }
}

// wave tile 32(M) x 64(N): acc 8 v8f = 64 VGPRs, frags 48 VGPRs -> stays under 256
__device__ __forceinline__ void compute_step(const bf16* Abuf, const bf16* Bbuf,
                                             int wm, int wn, int half, int l15,
                                             v8f acc[2][4]) {
    FragU af[2], bfr[4];
#pragma unroll
    for (int mf = 0; mf < 2; ++mf) load_a_frag(af[mf], Abuf, wm + mf * 16 + l15, half);
#pragma unroll
    for (int nf = 0; nf < 4; ++nf) load_b_frag(bfr[nf], Bbuf, wn + nf * 16 + l15, half);
#pragma unroll
    for (int mf = 0; mf < 2; ++mf)
#pragma unroll
        for (int nf = 0; nf < 4; ++nf)
            acc[mf][nf] = __builtin_amdgcn_wmma_f32_16x16x32_bf16(
                false, af[mf].v, false, bfr[nf].v, (short)0, acc[mf][nf], false, false);
}

// ---------------------------------------------------------------- GEMM1: H = gelu(X*W1 + b1)
__global__ __launch_bounds__(256)
void mod_gemm1_kernel(const bf16* __restrict__ A, const bf16* __restrict__ Bt,
                      const float* __restrict__ bias, bf16* __restrict__ H) {
    __shared__ __align__(16) bf16 Ab[2 * BM * ATS];
    __shared__ __align__(16) bf16 Bb[2 * BN * BTS];
    const int t    = threadIdx.x;
    const int lane = t & 31;
    const int wave = t >> 5;
    const int wm   = (wave & 3) * 32;
    const int wn   = (wave >> 2) * 64;
    const int m0   = blockIdx.y * BM;
    const int n0   = blockIdx.x * BN;
    const int half = lane >> 4;
    const int l15  = lane & 15;

    v8f acc[2][4] = {};

    stage_tiles(A, D_, Bt, D_, Ab, Bb, m0, n0, 0, t);
    const int NK = D_ / BK;
    for (int kt = 0; kt < NK; ++kt) {
        int cur = kt & 1;
        if (kt + 1 < NK) {
            stage_tiles(A, D_, Bt, D_, Ab + (cur ^ 1) * BM * ATS, Bb + (cur ^ 1) * BN * BTS,
                        m0, n0, (kt + 1) * BK, t);
            WAIT_ASYNC(4);
        } else {
            WAIT_ASYNC(0);
        }
        __syncthreads();
        compute_step(Ab + cur * BM * ATS, Bb + cur * BN * BTS, wm, wn, half, l15, acc);
        __syncthreads();
    }

#pragma unroll
    for (int mf = 0; mf < 2; ++mf)
#pragma unroll
        for (int nf = 0; nf < 4; ++nf) {
            int n = n0 + wn + nf * 16 + l15;
            float bv = bias[n];
#pragma unroll
            for (int r = 0; r < 8; ++r) {
                int m = m0 + wm + mf * 16 + half * 8 + r;
                float x = acc[mf][nf][r] + bv;
                float g = 0.5f * x * (1.f + tanhf(0.7978845608f * (x + 0.044715f * x * x * x)));
                H[(size_t)m * F_ + n] = (bf16)g;
            }
        }
}

// --------------------------- GEMM2: out[b,idx,:] = hidden + w*(H*W2 + b2), scatter epilogue
__global__ __launch_bounds__(256)
void mod_gemm2_kernel(const bf16* __restrict__ A, const bf16* __restrict__ Bt,
                      const float* __restrict__ bias, const int* __restrict__ top_idx,
                      const float* __restrict__ wgt, const float* __restrict__ hidden,
                      float* __restrict__ out) {
    __shared__ __align__(16) bf16 Ab[2 * BM * ATS];
    __shared__ __align__(16) bf16 Bb[2 * BN * BTS];
    const int t    = threadIdx.x;
    const int lane = t & 31;
    const int wave = t >> 5;
    const int wm   = (wave & 3) * 32;
    const int wn   = (wave >> 2) * 64;
    const int m0   = blockIdx.y * BM;
    const int n0   = blockIdx.x * BN;
    const int half = lane >> 4;
    const int l15  = lane & 15;

    v8f acc[2][4] = {};

    stage_tiles(A, F_, Bt, F_, Ab, Bb, m0, n0, 0, t);
    const int NK = F_ / BK;
    for (int kt = 0; kt < NK; ++kt) {
        int cur = kt & 1;
        if (kt + 1 < NK) {
            stage_tiles(A, F_, Bt, F_, Ab + (cur ^ 1) * BM * ATS, Bb + (cur ^ 1) * BN * BTS,
                        m0, n0, (kt + 1) * BK, t);
            WAIT_ASYNC(4);
        } else {
            WAIT_ASYNC(0);
        }
        __syncthreads();
        compute_step(Ab + cur * BM * ATS, Bb + cur * BN * BTS, wm, wn, half, l15, acc);
        __syncthreads();
    }

    // indices are unique per batch -> each selected row owned by one tile row, no atomics
#pragma unroll
    for (int mf = 0; mf < 2; ++mf)
#pragma unroll
        for (int nf = 0; nf < 4; ++nf) {
            int n = n0 + wn + nf * 16 + l15;
            float bv = bias[n];
#pragma unroll
            for (int r = 0; r < 8; ++r) {
                int m = m0 + wm + mf * 16 + half * 8 + r;
                int b = m >> 11;                 // / KCAP
                int s = top_idx[m];
                float w = wgt[m];
                size_t off = ((size_t)b * S_ + s) * D_ + n;
                out[off] = hidden[off] + w * (acc[mf][nf][r] + bv);
            }
        }
}

// ---------------------------------------------------------------- host launcher
extern "C" void kernel_launch(void* const* d_in, const int* in_sizes, int n_in,
                              void* d_out, int out_size, void* d_ws, size_t ws_size,
                              hipStream_t stream) {
    const float* hidden   = (const float*)d_in[0];
    const float* w_router = (const float*)d_in[1];
    const float* W1       = (const float*)d_in[2];
    const float* b1       = (const float*)d_in[3];
    const float* W2       = (const float*)d_in[4];
    const float* b2       = (const float*)d_in[5];
    float* out = (float*)d_out;

    char* ws = (char*)d_ws;
    size_t off = 0;
    auto take = [&](size_t bytes) -> void* {
        off = (off + 255) & ~(size_t)255;
        void* p = ws + off;
        off += bytes;
        return p;
    };
    float* scores  = (float*)take((size_t)B_ * S_ * sizeof(float));
    int*   top_idx = (int*)  take((size_t)M_ * sizeof(int));
    float* weights = (float*)take((size_t)M_ * sizeof(float));
    bf16*  Xb      = (bf16*) take((size_t)M_ * D_ * sizeof(bf16));
    bf16*  W1t     = (bf16*) take((size_t)D_ * F_ * sizeof(bf16));  // [F_][D_]
    bf16*  W2t     = (bf16*) take((size_t)F_ * D_ * sizeof(bf16));  // [D_][F_]
    bf16*  H       = (bf16*) take((size_t)M_ * F_ * sizeof(bf16));
    (void)ws_size; (void)in_sizes; (void)n_in; (void)out_size;

    // transpose-convert weights to N-major bf16 (B tiles become pure b128 copies)
    mod_cvt_t_kernel<<<dim3(F_ / 32, D_ / 32), 256, 0, stream>>>(W1, W1t, D_, F_);
    mod_cvt_t_kernel<<<dim3(D_ / 32, F_ / 32), 256, 0, stream>>>(W2, W2t, F_, D_);

    // router scores: one wave per token
    mod_scores_kernel<<<(B_ * S_) / 8, 256, 0, stream>>>(hidden, w_router, scores);

    // exact per-batch top-k via LDS bitonic sort
    mod_topk_kernel<<<B_, 1024, 0, stream>>>(scores, top_idx, weights);

    // gather selected tokens to bf16
    mod_gather_kernel<<<M_, 256, 0, stream>>>(hidden, top_idx, Xb);

    // bypass path: out = hidden (selected rows overwritten by gemm2 epilogue)
    (void)hipMemcpyAsync(out, hidden, (size_t)B_ * S_ * D_ * sizeof(float),
                         hipMemcpyDeviceToDevice, stream);

    // GEMM1 + GELU: (8192 x 1024) x (1024 x 4096)
    mod_gemm1_kernel<<<dim3(F_ / BN, M_ / BM), 256, 0, stream>>>(Xb, W1t, b1, H);

    // GEMM2 + weighted scatter: (8192 x 4096) x (4096 x 1024)
    mod_gemm2_kernel<<<dim3(D_ / BN, M_ / BM), 256, 0, stream>>>(H, W2t, b2, top_idx,
                                                                 weights, hidden, out);
}
